// DCG_payoff_78743930405296
// MI455X (gfx1250) — matrix operand order repaired and checked by reading, not compiled, not run on hardware
//
#include <hip/hip_runtime.h>
#include <hip/hip_bf16.h>

// Problem constants (from reference)
#define B_  256
#define N_  32
#define DH_ 128
#define E_  496
#define A_  12
#define R_  4

typedef _Float16 v4h  __attribute__((ext_vector_type(4)));
typedef _Float16 v8h  __attribute__((ext_vector_type(8)));
typedef _Float16 v16h __attribute__((ext_vector_type(16)));
typedef float    v4f  __attribute__((ext_vector_type(4)));
typedef float    v8f  __attribute__((ext_vector_type(8)));

__device__ __forceinline__ v16h cat8(v8h a, v8h b) {
  return __builtin_shufflevector(a, b, 0,1,2,3,4,5,6,7,8,9,10,11,12,13,14,15);
}

// Split fp32 into f16 hi/lo so hi*hi + hi*lo + lo*hi ~= fp32 product.
__device__ __forceinline__ void splitf(float x, _Float16* hi, _Float16* lo) {
  _Float16 h = (_Float16)x;
  *hi = h;
  *lo = (_Float16)(x - (float)h);
}

__device__ __forceinline__ void splitf4(v4f x, v4h* hi, v4h* lo) {
  v4h h, l;
  #pragma unroll
  for (int q = 0; q < 4; ++q) {
    _Float16 hh = (_Float16)x[q];
    h[q] = hh;
    l[q] = (_Float16)(x[q] - (float)hh);
  }
  *hi = h; *lo = l;
}

// A-fragment (16x32 f16, M x K) from an LDS plane laid out [row][k], row stride RS halves.
// Contiguous k order used here (B matches): lanes 0-15: M=lane, K = c*32 + 0..15; lanes 16-31: M=lane-16, K = c*32 + 16..31
template<int RS>
__device__ __forceinline__ v16h load_a_frag(const _Float16* plane, int mrow0, int chunk, int lane) {
  int row = mrow0 + (lane & 15);
  int kb  = chunk * 32 + ((lane >> 4) & 1) * 16;
  const _Float16* p = plane + row * RS + kb;
  v8h lo = *(const v8h*)(p);
  v8h hi = *(const v8h*)(p + 8);
  return cat8(lo, hi);
}

// B-fragment (32x16 f16, K x N) from a global plane laid out [n][k] (k stride = KS halves).
// Contiguous k order used here (A matches): lanes 0-15: N=lane, K = c*32 + 0..15; lanes 16-31: N=lane-16, K = c*32 + 16..31
__device__ __forceinline__ v16h load_b_frag(const _Float16* plane, int ncol0, int KS, int chunk, int lane) {
  int n  = ncol0 + (lane & 15);
  int kb = chunk * 32 + ((lane >> 4) & 1) * 16;
  return *(const v16h*)(plane + n * KS + kb);
}

// fp32-accurate product via three f16 WMMAs (skip lo*lo term).
__device__ __forceinline__ v8f wmma3(v16h ah, v16h al, v16h bh, v16h bl, v8f acc) {
  acc = __builtin_amdgcn_wmma_f32_16x16x32_f16(false, al, false, bh, (short)0, acc, false, false);
  acc = __builtin_amdgcn_wmma_f32_16x16x32_f16(false, ah, false, bl, (short)0, acc, false, false);
  acc = __builtin_amdgcn_wmma_f32_16x16x32_f16(false, ah, false, bh, (short)0, acc, false, false);
  return acc;
}

// ---------------------------------------------------------------------------
// Kernel 0: convert weights to hi/lo f16 planes in [n][k] order.
//   Wcat[512][128]: Wcat[j][k] = W1[j][k] (j<256) ; Wcat[256+j][k] = W1[j][128+k]
//   W2p [96][256] : row-major W2 as-is.
// ---------------------------------------------------------------------------
__global__ void __launch_bounds__(256)
wconv_kernel(const float* __restrict__ W1, const float* __restrict__ W2,
             _Float16* __restrict__ wcatHi, _Float16* __restrict__ wcatLo,
             _Float16* __restrict__ w2Hi,   _Float16* __restrict__ w2Lo) {
  int idx = blockIdx.x * 256 + threadIdx.x;
  if (idx < 512 * 128) {
    int n = idx >> 7, k = idx & 127;
    float x = (n < 256) ? W1[n * 256 + k] : W1[(n - 256) * 256 + 128 + k];
    splitf(x, &wcatHi[idx], &wcatLo[idx]);
  } else if (idx < 512 * 128 + 96 * 256) {
    int j = idx - 512 * 128;
    splitf(W2[j], &w2Hi[j], &w2Lo[j]);
  }
}

// ---------------------------------------------------------------------------
// Kernel 1: Y[b*32+n][0:512] = [W1a @ h[b,n] ; W1b @ h[b,n]]
// GEMM M=8192, K=128, Nout=512 via f16x3-split WMMA. One 16-row M-tile / block.
// ---------------------------------------------------------------------------
__global__ void __launch_bounds__(128)
yproj_kernel(const float* __restrict__ hidden,
             const _Float16* __restrict__ wcatHi, const _Float16* __restrict__ wcatLo,
             float* __restrict__ Y) {
  __shared__ __align__(32) _Float16 hHi[16 * 128];
  __shared__ __align__(32) _Float16 hLo[16 * 128];

  int mtile = blockIdx.x;          // 0..511 ; row block = rows [mtile*16, +16)
  int t = threadIdx.x;
  const v4f* src = (const v4f*)(hidden + (size_t)mtile * 16 * 128);

  // Stage 16x128 fp32 rows -> hi/lo f16 planes (vectorized: 512 float4 items).
  #pragma unroll
  for (int i = 0; i < 4; ++i) {
    int idx = t + i * 128;
    v4h h, l;
    splitf4(src[idx], &h, &l);
    *(v4h*)(&hHi[idx * 4]) = h;
    *(v4h*)(&hLo[idx * 4]) = l;
  }
  __syncthreads();

  int wave = t >> 5, lane = t & 31;
  for (int nt = wave; nt < 32; nt += 4) {           // wave-uniform: EXEC all ones
    v8f acc = {};
    #pragma unroll
    for (int c = 0; c < 4; ++c) {
      v16h ah = load_a_frag<128>(hHi, 0, c, lane);
      v16h al = load_a_frag<128>(hLo, 0, c, lane);
      v16h bh = load_b_frag(wcatHi, nt * 16, 128, c, lane);
      v16h bl = load_b_frag(wcatLo, nt * 16, 128, c, lane);
      acc = wmma3(ah, al, bh, bl, acc);
    }
    // C layout: VGPR r -> row (r + 8*(lane>>4)), col = lane&15
    int col   = nt * 16 + (lane & 15);
    int rbase = mtile * 16 + ((lane >> 4) & 1) * 8;
    #pragma unroll
    for (int r = 0; r < 8; ++r)
      Y[(size_t)(rbase + r) * 512 + col] = acc[r];
  }
}

// ---------------------------------------------------------------------------
// Kernel 2: per (b, 32-edge group): gather Y, +b1, ReLU, split -> LDS act tiles,
// layer-2 GEMM (K=256 -> 96) via WMMA for both sides, then payoff contraction.
// ---------------------------------------------------------------------------
__global__ void __launch_bounds__(256)
edge_kernel(const float* __restrict__ Y,
            const _Float16* __restrict__ w2Hi, const _Float16* __restrict__ w2Lo,
            const float* __restrict__ b1, const float* __restrict__ b2,
            const int* __restrict__ ef, const int* __restrict__ et,
            float* __restrict__ out) {
  __shared__ __align__(32) _Float16 actHi[32 * 256];   // 16 KB
  __shared__ __align__(32) _Float16 actLo[32 * 256];   // 16 KB
  __shared__ float pLds[2 * 32 * 96];                  // 24 KB
  __shared__ int   eFrom[32], eTo[32];

  int b  = blockIdx.y;
  int e0 = blockIdx.x * 32;
  int t  = threadIdx.x;
  int wave = t >> 5, lane = t & 31;

  // Warm WGP$/L2 with the W2 planes this block will stream through WMMA.
  {
    const char* p0 = (const char*)w2Hi;
    __builtin_prefetch(p0 + t * 384, 0, 0);           // 96 KB hi+lo / 256 threads
  }

  // Cache the 32 (clamped) edge node indices once.
  if (t < 32) {
    int e = e0 + t;
    int ec = e < E_ ? e : E_ - 1;
    eFrom[t] = ef[ec];
    eTo[t]   = et[ec];
  }
  // Per-thread fixed k-slice: k4 = t&63 -> k = [4*k4, 4*k4+4). Hoist b1 vec.
  int k4 = t & 63;
  v4f b1v = *(const v4f*)(b1 + k4 * 4);
  __syncthreads();

  for (int s = 0; s < 2; ++s) {
    // ---- stage activations for side s: 32 edges x 256 hidden (float4 items) ----
    #pragma unroll
    for (int i = 0; i < 8; ++i) {
      int idx = t + i * 256;               // 2048 float4 items
      int el  = idx >> 6;                  // edge-local 0..31
      int na = (s == 0) ? eFrom[el] : eTo[el];
      int nb = (s == 0) ? eTo[el]   : eFrom[el];
      v4f va = *(const v4f*)(Y + (size_t)(b * 32 + na) * 512 + k4 * 4);
      v4f vb = *(const v4f*)(Y + (size_t)(b * 32 + nb) * 512 + 256 + k4 * 4);
      v4f v = va + vb + b1v;
      #pragma unroll
      for (int q = 0; q < 4; ++q) v[q] = v[q] > 0.f ? v[q] : 0.f;  // ReLU
      v4h h, l;
      splitf4(v, &h, &l);
      *(v4h*)(&actHi[idx * 4]) = h;
      *(v4h*)(&actLo[idx * 4]) = l;
    }
    __syncthreads();

    // ---- layer-2 WMMA: 12 jobs = 2 M-tiles x 6 N-tiles over 8 waves ----
    for (int job = wave; job < 12; job += 8) {   // wave-uniform
      int mt = job / 6, ntile = job % 6;
      float bias = b2[ntile * 16 + (lane & 15)];
      v8f acc;
      #pragma unroll
      for (int r = 0; r < 8; ++r) acc[r] = bias;
      #pragma unroll
      for (int c = 0; c < 8; ++c) {
        v16h ah = load_a_frag<256>(actHi, mt * 16, c, lane);
        v16h al = load_a_frag<256>(actLo, mt * 16, c, lane);
        v16h bh = load_b_frag(w2Hi, ntile * 16, 256, c, lane);
        v16h bl = load_b_frag(w2Lo, ntile * 16, 256, c, lane);
        acc = wmma3(ah, al, bh, bl, acc);
      }
      int col   = ntile * 16 + (lane & 15);
      int rbase = mt * 16 + ((lane >> 4) & 1) * 8;
      float* pp = pLds + s * 32 * 96;
      #pragma unroll
      for (int r = 0; r < 8; ++r)
        pp[(rbase + r) * 96 + col] = acc[r];
    }
    __syncthreads();   // also protects act LDS before side-1 restage
  }

  // ---- payoff: out[b,e,i,j] = 0.5*(sum_r p0[r,0,i]p0[r,1,j] + p1[r,0,j]p1[r,1,i])
  // One (edge, i-row) per work item: reuse p0i/p1i quads across all 12 j's,
  // and emit 12 contiguous output floats (vectorizable stores).
  #pragma unroll
  for (int i = 0; i < 2; ++i) {
    int idx = t + i * 256;                 // 384 = 32 edges * 12 rows
    if (idx < 384) {
      int el = idx / 12, ii = idx % 12;
      int e = e0 + el;
      if (e < E_) {
        const float* p0 = pLds + el * 96;
        const float* p1 = pLds + 32 * 96 + el * 96;
        float p0i[R_], p1i[R_];
        #pragma unroll
        for (int r = 0; r < R_; ++r) {
          p0i[r] = p0[r * 24 + ii];        // p0[r, 0, ii]
          p1i[r] = p1[r * 24 + 12 + ii];   // p1[r, 1, ii]
        }
        float res[A_];
        #pragma unroll
        for (int jj = 0; jj < A_; ++jj) {
          float a = 0.f;
          #pragma unroll
          for (int r = 0; r < R_; ++r) {
            a += p0i[r] * p0[r * 24 + 12 + jj];   // payoffs[0][ii][jj]
            a += p1[r * 24 + jj] * p1i[r];        // payoffs[1][jj][ii]
          }
          res[jj] = 0.5f * a;
        }
        float* op = out + ((size_t)(b * E_ + e)) * 144 + ii * 12;
        #pragma unroll
        for (int jj = 0; jj < A_; ++jj) op[jj] = res[jj];
      }
    }
  }
}

// ---------------------------------------------------------------------------
extern "C" void kernel_launch(void* const* d_in, const int* in_sizes, int n_in,
                              void* d_out, int out_size, void* d_ws, size_t ws_size,
                              hipStream_t stream) {
  const float* hidden = (const float*)d_in[0];
  const float* W1     = (const float*)d_in[1];
  const float* b1     = (const float*)d_in[2];
  const float* W2     = (const float*)d_in[3];
  const float* b2     = (const float*)d_in[4];
  const int*   ef     = (const int*)d_in[5];
  const int*   et     = (const int*)d_in[6];
  float* out          = (float*)d_out;

  // Workspace layout
  char* ws = (char*)d_ws;
  float*    Y      = (float*)ws;                                  // 8192*512*4 = 16 MB
  _Float16* wcatHi = (_Float16*)(ws + (size_t)16 * 1024 * 1024);  // 512*128
  _Float16* wcatLo = wcatHi + 512 * 128;
  _Float16* w2Hi   = wcatLo + 512 * 128;                          // 96*256 (lo follows -> prefetch covers both)
  _Float16* w2Lo   = w2Hi + 96 * 256;

  // 0) weight conversion: 512*128 + 96*256 = 90112 elements
  wconv_kernel<<<(90112 + 255) / 256, 256, 0, stream>>>(W1, W2, wcatHi, wcatLo, w2Hi, w2Lo);

  // 1) node projection Y = [W1a@h ; W1b@h] : 512 M-tiles
  yproj_kernel<<<512, 128, 0, stream>>>(hidden, wcatHi, wcatLo, Y);

  // 2) per-edge MLP layer 2 + payoff: 16 edge groups x 256 batches
  dim3 grid((E_ + 31) / 32, B_);
  edge_kernel<<<grid, 256, 0, stream>>>(Y, w2Hi, w2Lo, b1, b2, ef, et, out);
}
